// CantorMoELayer_57174604644751
// MI455X (gfx1250) — matrix operand order repaired and checked by reading, not compiled, not run on hardware
//
#include <hip/hip_runtime.h>
#include <math.h>

#define E 16
#define P 4096
#define BB 8
#define DFULL 2048
#define EDIM 128
#define SLICE 128
#define M 256
#define WIN 3
#define GATE_H 32

typedef __attribute__((ext_vector_type(16))) __bf16 v16bf;
typedef __attribute__((ext_vector_type(2)))  __bf16 bf16x2;
typedef __attribute__((ext_vector_type(8)))  float  v8f;

union AFrag { v16bf v; bf16x2 p[8]; unsigned u[8]; };
union BFrag { v16bf v; float4 f4[2]; unsigned u[8]; };

// hardware f32->bf16 (RNE fptrunc, lowers to packed cvt), two packed per dword
__device__ __forceinline__ unsigned pack_bf16(float a, float b) {
  bf16x2 p;
  p.x = (__bf16)a;
  p.y = (__bf16)b;
  unsigned r;
  __builtin_memcpy(&r, &p, 4);
  return r;
}
__device__ __forceinline__ unsigned short bf16_1(float a) {
  __bf16 h = (__bf16)a;
  unsigned short r;
  __builtin_memcpy(&r, &h, 2);
  return r;
}

// ---------------------------------------------------------------------------
// Kernel 1: per-expert token index compaction (order-preserving, matches
// jnp.nonzero(size=M, fill_value=0)). Trivial work: 16 scans of 4096.
// ---------------------------------------------------------------------------
__global__ void route_kernel(const float* __restrict__ fp, int* __restrict__ idx) {
  int e = blockIdx.x;
  if (threadIdx.x != 0) return;
  float lo = (float)e / (float)E;
  float hi = (float)(e + 1) / (float)E;
  int cnt = 0;
  for (int p = 0; p < P; ++p) {
    float f = fp[p];
    bool m = (f >= lo) && ((e == E - 1) ? (f <= hi) : (f < hi));
    if (m && cnt < M) idx[e * M + cnt++] = p;
  }
  while (cnt < M) idx[e * M + cnt++] = 0;
}

// ---------------------------------------------------------------------------
// Kernel 2: fold normalized penta directions into wq/wk:
//   wqd[e][v][c] = sum_d wq[e][c][d] * penta_hat[e][v][d]
// (removes the full 256x128x128 Q/K GEMMs), and pre-convert wv to bf16 once
// so the expert GEMM's B operand needs no per-wave conversion.
// ---------------------------------------------------------------------------
__global__ void fold_kernel(const float* __restrict__ wq, const float* __restrict__ wk,
                            const float* __restrict__ wv, const float* __restrict__ penta,
                            float* __restrict__ wqd, float* __restrict__ wkd,
                            unsigned* __restrict__ wvbf) {
  __shared__ float pen[5 * EDIM];
  __shared__ float rn[5];
  int e = blockIdx.x, t = threadIdx.x;
  for (int i = t; i < 5 * EDIM; i += 128) pen[i] = penta[(size_t)e * 5 * EDIM + i];
  __syncthreads();
  if (t < 5) {
    float s = 0.f;
    for (int d = 0; d < EDIM; ++d) { float x = pen[t * EDIM + d]; s += x * x; }
    rn[t] = 1.0f / fmaxf(sqrtf(s), 1e-12f);
  }
  __syncthreads();
  float aq[5] = {0, 0, 0, 0, 0}, ak[5] = {0, 0, 0, 0, 0};
  const float* wqr = wq + ((size_t)e * SLICE + t) * EDIM;
  const float* wkr = wk + ((size_t)e * SLICE + t) * EDIM;
  for (int d = 0; d < EDIM; ++d) {
    float q = wqr[d], k = wkr[d];
#pragma unroll
    for (int v = 0; v < 5; ++v) { float pv = pen[v * EDIM + d]; aq[v] += q * pv; ak[v] += k * pv; }
  }
#pragma unroll
  for (int v = 0; v < 5; ++v) {
    wqd[((size_t)e * 5 + v) * SLICE + t] = aq[v] * rn[v];
    wkd[((size_t)e * 5 + v) * SLICE + t] = ak[v] * rn[v];
  }
  // one-time wv f32 -> bf16 (packed pairs)
  const float2* src = (const float2*)(wv + (size_t)e * SLICE * EDIM);
  unsigned* dst = wvbf + (size_t)e * SLICE * (EDIM / 2);
  for (int i = t; i < SLICE * EDIM / 2; i += 128) {
    float2 f = src[i];
    dst[i] = pack_bf16(f.x, f.y);
  }
}

// ---------------------------------------------------------------------------
// Kernel 3: per (e,b,64-row tile): gather slice, gate MLP (exact gelu),
// scale, Qa/Ka via folded 128x5 dots, V = feats @ wv via WMMA bf16.
// ---------------------------------------------------------------------------
__global__ __launch_bounds__(128) void expert_kernel(
    const float* __restrict__ tokens, const float* __restrict__ alpha,
    const float* __restrict__ gw1, const float* __restrict__ gb1,
    const float* __restrict__ gw2, const float* __restrict__ gb2,
    const unsigned* __restrict__ wvbf,
    const float* __restrict__ wqd, const float* __restrict__ wkd,
    const int* __restrict__ idx,
    float* __restrict__ Qa, float* __restrict__ Ka,
    unsigned short* __restrict__ Vbf) {
  __shared__ float sF[64 * 132];   // 64 rows x 128 cols, stride 132 (bank-safe)
  __shared__ int sTok[64];

  int bid = blockIdx.x;
  int rt = bid & 3, b = (bid >> 2) & 7, e = bid >> 5;
  int tid = threadIdx.x;

  if (tid < 64) sTok[tid] = idx[e * M + rt * 64 + tid];
  __syncthreads();

  // gather (coalesced b128 loads of the expert's 128-col slice)
  for (int i = tid; i < 64 * 32; i += 128) {
    int r = i >> 5, c4 = i & 31;
    const float4 vv = ((const float4*)(tokens + ((size_t)b * P + sTok[r]) * DFULL + (size_t)e * SLICE))[c4];
    float* dst = &sF[r * 132 + c4 * 4];
    dst[0] = vv.x; dst[1] = vv.y; dst[2] = vv.z; dst[3] = vv.w;
  }
  __syncthreads();

  // gate MLP + in-place row scale (one thread per row)
  if (tid < 64) {
    int r = tid;
    float h[GATE_H];
#pragma unroll
    for (int j = 0; j < GATE_H; ++j) h[j] = gb1[e * GATE_H + j];
    const float* g1 = gw1 + (size_t)e * SLICE * GATE_H;
    for (int c = 0; c < SLICE; ++c) {
      float f = sF[r * 132 + c];
      const float* g1r = g1 + c * GATE_H;
#pragma unroll
      for (int j = 0; j < GATE_H; ++j) h[j] += f * g1r[j];
    }
    float g = gb2[e];
    const float* g2 = gw2 + e * GATE_H;
#pragma unroll
    for (int j = 0; j < GATE_H; ++j) {
      float x = h[j];
      g += 0.5f * x * (1.0f + erff(x * 0.70710678118654752f)) * g2[j];
    }
    g = 1.0f / (1.0f + expf(-g));
    float aw = 1.0f / (1.0f + expf(-alpha[e]));
    float s = g * aw + (1.0f - aw);
    for (int c = 0; c < SLICE; ++c) sF[r * 132 + c] *= s;
  }
  __syncthreads();

  // Qa/Ka via folded projections: threads 0..63 -> Qa, 64..127 -> Ka
  {
    int r = tid & 63;
    const float* wd = (tid < 64) ? wqd : wkd;
    float* dst = (tid < 64) ? Qa : Ka;
    float acc5[5] = {0, 0, 0, 0, 0};
    for (int c = 0; c < SLICE; ++c) {
      float f = sF[r * 132 + c];
#pragma unroll
      for (int v = 0; v < 5; ++v) acc5[v] += f * wd[((size_t)e * 5 + v) * SLICE + c];
    }
    int m = rt * 64 + r;
#pragma unroll
    for (int v = 0; v < 5; ++v)
      dst[(((size_t)e * BB + b) * 5 + v) * M + m] = acc5[v];
  }
  // sF is read-only from here: no barrier needed before the GEMM.

  // V = feats @ wv : each wave owns a 16x128 output tile, K = 128 (4 steps).
  int w = tid >> 5, l = tid & 31;
  int arow = w * 16 + (l & 15);
  int kbase = (l >> 4) * 8;

  v8f acc[8];
  v8f zero = {0.f, 0.f, 0.f, 0.f, 0.f, 0.f, 0.f, 0.f};
#pragma unroll
  for (int nt = 0; nt < 8; ++nt) acc[nt] = zero;

  for (int kk = 0; kk < 4; ++kk) {
    AFrag A;
#pragma unroll
    for (int j = 0; j < 8; ++j) {
      int koff = kbase + ((j < 4) ? 2 * j : 2 * j + 8);
      const float2 fv = *(const float2*)&sF[arow * 132 + kk * 32 + koff];
      A.p[j].x = (__bf16)fv.x;
      A.p[j].y = (__bf16)fv.y;
    }
    const unsigned short* wcol = (const unsigned short*)wvbf + (size_t)e * SLICE * EDIM + (l & 15);
#pragma unroll
    for (int nt = 0; nt < 8; ++nt) {
      BFrag Bf;
#pragma unroll
      for (int j = 0; j < 8; ++j) {
        int k0 = kk * 32 + kbase + ((j < 4) ? 2 * j : 2 * j + 8);
        Bf.u[j] = (unsigned)wcol[(size_t)k0 * EDIM + nt * 16] | ((unsigned)wcol[(size_t)(k0 + 1) * EDIM + nt * 16] << 16);
      }
      acc[nt] = __builtin_amdgcn_wmma_f32_16x16x32_bf16(false, A.v, false, Bf.v,
                                                        (short)0, acc[nt], false, false);
    }
  }

  // store V as bf16 (B operand of the attention GEMM)
#pragma unroll
  for (int nt = 0; nt < 8; ++nt) {
#pragma unroll
    for (int r = 0; r < 8; ++r) {
      int mloc = rt * 64 + w * 16 + r + ((l >> 4) << 3);
      int d = nt * 16 + (l & 15);
      Vbf[(((size_t)e * BB + b) * M + mloc) * EDIM + d] = bf16_1(acc[nt][r]);
    }
  }
}

// ---------------------------------------------------------------------------
// Kernel 4: attention with fused fusion-weights:
//   P'[m,n] = sum_v w[v] * softmax_n(q_v[m]*k_v[n]/T);  out = P' @ Vn
// Rank-1 scores -> row max is q*kmax or q*kmin (O(1) per row).
// ---------------------------------------------------------------------------
__global__ __launch_bounds__(256) void attn_kernel(
    const float* __restrict__ Qa, const float* __restrict__ Ka,
    const unsigned short* __restrict__ Vbf,
    const int* __restrict__ idx, const int* __restrict__ routes,
    const float* __restrict__ temperature, const float* __restrict__ fusion_w,
    float* __restrict__ out) {
  __shared__ float sKn[5 * 768];
  __shared__ float sQ[5 * 128];
  __shared__ float sC[5 * 128];
  __shared__ float sZ[5 * 128];
  __shared__ int sTok[128];
  __shared__ float sMx[5], sMn[5];

  int bid = blockIdx.x;
  int half = bid & 1, b = (bid >> 1) & 7, i = bid >> 4;
  int tid = threadIdx.x;
  int nb0 = routes[i * WIN + 0], nb1 = routes[i * WIN + 1], nb2 = routes[i * WIN + 2];
  float invT = 1.0f / temperature[0];

  // fusion-weight softmax, held in registers by every thread
  float fw[5];
  {
    float mx = -1e30f;
#pragma unroll
    for (int v = 0; v < 5; ++v) { fw[v] = fusion_w[v]; mx = fmaxf(mx, fw[v]); }
    float s = 0.f;
#pragma unroll
    for (int v = 0; v < 5; ++v) { fw[v] = expf(fw[v] - mx); s += fw[v]; }
#pragma unroll
    for (int v = 0; v < 5; ++v) fw[v] /= s;
  }

  // stage neighbor K projections and this block's Q rows
  for (int t = tid; t < 5 * 768; t += 256) {
    int v = t / 768, n = t % 768;
    int jn = n >> 8, mloc = n & 255;
    int ne = (jn == 0) ? nb0 : ((jn == 1) ? nb1 : nb2);
    sKn[t] = Ka[(((size_t)ne * BB + b) * 5 + v) * M + mloc];
  }
  for (int t = tid; t < 5 * 128; t += 256) {
    int v = t >> 7, r = t & 127;
    sQ[t] = Qa[(((size_t)i * BB + b) * 5 + v) * M + half * 128 + r];
  }
  if (tid < 128) sTok[tid] = idx[i * M + half * 128 + tid];
  __syncthreads();

  int w = tid >> 5, l = tid & 31;

  // per-view kmin/kmax (waves 0..4, wave32 shuffle reduction)
  if (w < 5) {
    float mx = -1e30f, mn = 1e30f;
    for (int n = l; n < 768; n += 32) { float k = sKn[w * 768 + n]; mx = fmaxf(mx, k); mn = fminf(mn, k); }
#pragma unroll
    for (int o = 16; o > 0; o >>= 1) {
      mx = fmaxf(mx, __shfl_xor(mx, o, 32));
      mn = fminf(mn, __shfl_xor(mn, o, 32));
    }
    if (l == 0) { sMx[w] = mx; sMn[w] = mn; }
  }
  __syncthreads();

  // Z pass: per (view,row) softmax denominator; fold w[v]/Z into one scalar
  for (int t = tid; t < 5 * 128; t += 256) {
    int v = t >> 7, r = t & 127;
    float q = sQ[v * 128 + r];
    float cmx = ((q >= 0.f) ? q * sMx[v] : q * sMn[v]) * invT;
    float z = 0.f;
    const float* kn = &sKn[v * 768];
    for (int n = 0; n < 768; ++n) z += __expf(q * kn[n] * invT - cmx);
    sC[t] = cmx;
    sZ[t] = fw[v] / z;
  }
  __syncthreads();

  // GEMM: each wave 16 rows x 128 cols, K = 768 (24 steps of 32)
  int rowl = w * 16 + (l & 15);
  float qv[5], cv[5], zv[5];
#pragma unroll
  for (int v = 0; v < 5; ++v) {
    qv[v] = sQ[v * 128 + rowl] * invT;
    cv[v] = sC[v * 128 + rowl];
    zv[v] = sZ[v * 128 + rowl];
  }
  int kbase = (l >> 4) * 8;

  v8f acc[8];
  v8f zero = {0.f, 0.f, 0.f, 0.f, 0.f, 0.f, 0.f, 0.f};
#pragma unroll
  for (int nt = 0; nt < 8; ++nt) acc[nt] = zero;

  for (int kk = 0; kk < 24; ++kk) {
    AFrag A;
#pragma unroll
    for (int j = 0; j < 8; ++j) {
      int koff = kbase + ((j < 4) ? 2 * j : 2 * j + 8);
      int n0 = kk * 32 + koff;
      float p0 = 0.f, p1 = 0.f;
#pragma unroll
      for (int v = 0; v < 5; ++v) {
        const float* kn = &sKn[v * 768];
        p0 += __expf(qv[v] * kn[n0]     - cv[v]) * zv[v];
        p1 += __expf(qv[v] * kn[n0 + 1] - cv[v]) * zv[v];
      }
      A.p[j].x = (__bf16)p0;
      A.p[j].y = (__bf16)p1;
    }
#pragma unroll
    for (int nt = 0; nt < 8; ++nt) {
      BFrag Bf;
#pragma unroll
      for (int j = 0; j < 8; ++j) {
        int n = kk * 32 + kbase + ((j < 4) ? 2 * j : 2 * j + 8);
        int jn = n >> 8, mloc = n & 255;
        int ne = (jn == 0) ? nb0 : ((jn == 1) ? nb1 : nb2);
        const unsigned short* vrow = Vbf + (((size_t)ne * BB + b) * M + mloc) * EDIM + nt * 16 + (l & 15);
        Bf.u[j] = (unsigned)vrow[0] | ((unsigned)vrow[EDIM] << 16);
      }
      acc[nt] = __builtin_amdgcn_wmma_f32_16x16x32_bf16(false, A.v, false, Bf.v,
                                                        (short)0, acc[nt], false, false);
    }
  }

  // scatter to out[b, idx[i][m], :] (expert partition covers every token once)
#pragma unroll
  for (int nt = 0; nt < 8; ++nt) {
#pragma unroll
    for (int r = 0; r < 8; ++r) {
      int rl = w * 16 + r + ((l >> 4) << 3);
      int tok = sTok[rl];
      out[((size_t)b * P + tok) * EDIM + nt * 16 + (l & 15)] = acc[nt][r];
    }
  }
}

// ---------------------------------------------------------------------------
extern "C" void kernel_launch(void* const* d_in, const int* in_sizes, int n_in,
                              void* d_out, int out_size, void* d_ws, size_t ws_size,
                              hipStream_t stream) {
  (void)in_sizes; (void)n_in; (void)out_size; (void)ws_size;
  const float* tokens       = (const float*)d_in[0];
  const float* fingerprints = (const float*)d_in[1];
  const float* alpha        = (const float*)d_in[2];
  const float* gw1          = (const float*)d_in[3];
  const float* gb1          = (const float*)d_in[4];
  const float* gw2          = (const float*)d_in[5];
  const float* gb2          = (const float*)d_in[6];
  const float* wq           = (const float*)d_in[7];
  const float* wk           = (const float*)d_in[8];
  const float* wv           = (const float*)d_in[9];
  const float* penta        = (const float*)d_in[10];
  const float* temperature  = (const float*)d_in[11];
  const float* fusion_w     = (const float*)d_in[12];
  const int*   routes       = (const int*)d_in[13];
  float* out = (float*)d_out;

  char* ws = (char*)d_ws;
  size_t off = 0;
  auto alloc = [&](size_t bytes) { size_t o = off; off += (bytes + 255) & ~(size_t)255; return o; };
  int* idx            = (int*)(ws + alloc((size_t)E * M * sizeof(int)));
  float* wqd          = (float*)(ws + alloc((size_t)E * 5 * SLICE * sizeof(float)));
  float* wkd          = (float*)(ws + alloc((size_t)E * 5 * SLICE * sizeof(float)));
  float* Qa           = (float*)(ws + alloc((size_t)E * BB * 5 * M * sizeof(float)));
  float* Ka           = (float*)(ws + alloc((size_t)E * BB * 5 * M * sizeof(float)));
  unsigned short* Vbf = (unsigned short*)(ws + alloc((size_t)E * BB * M * EDIM * sizeof(unsigned short)));
  unsigned* wvbf      = (unsigned*)(ws + alloc((size_t)E * SLICE * (EDIM / 2) * sizeof(unsigned)));

  route_kernel<<<E, 32, 0, stream>>>(fingerprints, idx);
  fold_kernel<<<E, 128, 0, stream>>>(wq, wk, wv, penta, wqd, wkd, wvbf);
  expert_kernel<<<E * BB * 4, 128, 0, stream>>>(tokens, alpha, gw1, gb1, gw2, gb2,
                                                wvbf, wqd, wkd, idx, Qa, Ka, Vbf);
  attn_kernel<<<E * BB * 2, 256, 0, stream>>>(Qa, Ka, Vbf, idx, routes,
                                              temperature, fusion_w, out);
}
